// BiAttentionLayer_7318624272769
// MI455X (gfx1250) — matrix operand order repaired and checked by reading, not compiled, not run on hardware
//
#include <hip/hip_runtime.h>
#include <hip/hip_bf16.h>
#include <cstdint>

#define B_N 16
#define JX_N 1024
#define JQ_N 128
#define D_N 512
#define TILE_X 128
#define KC 64
#define VERY_NEG (-1e30f)

typedef __attribute__((ext_vector_type(16))) _Float16     v16h;
typedef __attribute__((ext_vector_type(8)))  float        v8f;
typedef __attribute__((ext_vector_type(8)))  unsigned int v8u;
typedef __attribute__((ext_vector_type(4)))  unsigned int u32x4;
typedef __attribute__((ext_vector_type(8)))  int          i32x8;
typedef __attribute__((ext_vector_type(4)))  int          i32x4;

// Exact pointee type of the async-copy builtin (from the round-2 diagnostic):
typedef int v4i __attribute__((vector_size(4 * sizeof(int))));
typedef __attribute__((address_space(1))) v4i gv4i;   // global
typedef __attribute__((address_space(3))) v4i lv4i;   // LDS

union Frag { v16h h; v8u u; };

// ---- gfx1250 feature probes (compile-safe fallbacks everywhere) -----------
#if defined(__has_builtin)
#  if __has_builtin(__builtin_amdgcn_global_load_async_to_lds_b128)
#    define HAS_ASYNC_COPY 1
#  endif
#  if __has_builtin(__builtin_amdgcn_s_wait_asynccnt)
#    define HAS_WAIT_ASYNC 1
#  endif
#  if __has_builtin(__builtin_amdgcn_s_wait_tensorcnt)
#    define HAS_WAIT_TENSOR 1
#  endif
#  if __has_builtin(__builtin_amdgcn_tensor_load_to_lds)
#    define HAS_TDM 1
#  endif
#endif
#ifndef HAS_ASYNC_COPY
#  define HAS_ASYNC_COPY 0
#endif
#ifndef HAS_WAIT_ASYNC
#  define HAS_WAIT_ASYNC 0
#endif
#ifndef HAS_WAIT_TENSOR
#  define HAS_WAIT_TENSOR 0
#endif
#ifndef HAS_TDM
#  define HAS_TDM 0
#endif

#if HAS_WAIT_ASYNC
#  define WAIT_ASYNC(n) __builtin_amdgcn_s_wait_asynccnt(n)
#elif HAS_ASYNC_COPY
#  define WAIT_ASYNC(n) asm volatile("s_wait_asynccnt %0" ::"i"(n) : "memory")
#else
#  define WAIT_ASYNC(n) ((void)0)
#endif

#if HAS_WAIT_TENSOR
#  define WAIT_TENSOR() __builtin_amdgcn_s_wait_tensorcnt(0)
#else
#  define WAIT_TENSOR() asm volatile("s_wait_tensorcnt 0x0" ::: "memory")
#endif

// 16-byte global -> LDS copy; async engine when available.
__device__ __forceinline__ void copy16(const _Float16* g, _Float16* l) {
#if HAS_ASYNC_COPY
  __builtin_amdgcn_global_load_async_to_lds_b128(
      (gv4i*)(uintptr_t)g,
      (lv4i*)(__attribute__((address_space(3))) void*)l, 0, 0);
#else
  *reinterpret_cast<uint4*>(l) = *reinterpret_cast<const uint4*>(g);
#endif
}

// 16-bit A-matrix 16x32 fragment: VGPR j holds K = {k0, k0+1},
// k0 = 2j + 8*half (+8 if j>=4)   (ISA 7.12.2)
__device__ __forceinline__ int a_k0(int j, int half) {
  return 2 * j + 8 * half + ((j >= 4) ? 8 : 0);
}

// ---------------------------------------------------------------------------
// cvt_h: h16 = (f16) h        (pure bandwidth, 8 halfs / thread)
// ---------------------------------------------------------------------------
__global__ __launch_bounds__(256)
void biattn_cvt_h(const float* __restrict__ src, _Float16* __restrict__ dst) {
  const size_t i = ((size_t)blockIdx.x * 256 + threadIdx.x) * 8;
  const float4 a = *reinterpret_cast<const float4*>(src + i);
  const float4 b = *reinterpret_cast<const float4*>(src + i + 4);
  _Float16 o[8] = {(_Float16)a.x, (_Float16)a.y, (_Float16)a.z, (_Float16)a.w,
                   (_Float16)b.x, (_Float16)b.y, (_Float16)b.z, (_Float16)b.w};
  *reinterpret_cast<uint4*>(dst + i) = *reinterpret_cast<const uint4*>(o);
}

// ---------------------------------------------------------------------------
// cvt_u: B16[b,q,d] = f16(u*w_hu + w_h)  (fused s_h trick),
//        u16T[b,d,q] = f16(u)            (GEMM2 B tile, K-pairs contiguous)
// ---------------------------------------------------------------------------
__global__ __launch_bounds__(256)
void biattn_cvt_u(const float* __restrict__ umat, const float* __restrict__ wvec,
                  _Float16* __restrict__ B16, _Float16* __restrict__ u16T) {
  const int b = blockIdx.x;
  const int tid = threadIdx.x;
  const float* wh = wvec;
  const float* whu = wvec + 2 * D_N;
  const float* ub = umat + (size_t)b * JQ_N * D_N;
  for (int q = 0; q < JQ_N; ++q) {
    for (int d = tid; d < D_N; d += 256) {
      const float v = ub[(size_t)q * D_N + d];
      B16[((size_t)b * JQ_N + q) * D_N + d] = (_Float16)(v * whu[d] + wh[d]);
      u16T[((size_t)b * D_N + d) * JQ_N + q] = (_Float16)v;
    }
  }
}

// ---------------------------------------------------------------------------
// prep: su[b,q] = dot(u[b,q,:], w_u) + bias + (u_mask ? 0 : VERY_NEG)
// ---------------------------------------------------------------------------
__global__ __launch_bounds__(256)
void biattn_prep(const float* __restrict__ umat, const float* __restrict__ wvec,
                 const float* __restrict__ bias, const unsigned char* __restrict__ umask,
                 float* __restrict__ su) {
  const int b = blockIdx.x;
  const int tid = threadIdx.x;
  const int q = tid >> 1;
  const int hf = tid & 1;
  const float* wu = wvec + D_N;
  const float* ur = umat + (size_t)(b * JQ_N + q) * D_N;
  float s = 0.0f;
  const int d0 = hf * 256;
  for (int d = d0; d < d0 + 256; ++d) s += ur[d] * wu[d];
  s += __shfl_xor(s, 1, 32);
  if (hf == 0)
    su[b * JQ_N + q] = s + bias[0] + (umask[b * JQ_N + q] ? 0.0f : VERY_NEG);
}

// ---------------------------------------------------------------------------
// core: async double-buffered staging + WMMA GEMM1, softmax over q,
//       TDM/async-staged WMMA GEMM2 -> u_a; h_logits to ws
// ---------------------------------------------------------------------------
__global__ __launch_bounds__(256)
void biattn_core(const _Float16* __restrict__ h16, const _Float16* __restrict__ B16,
                 const _Float16* __restrict__ u16T,
                 const unsigned char* __restrict__ hmask, const float* __restrict__ su,
                 float* __restrict__ out_ua, float* __restrict__ hlog) {
  __shared__ _Float16 smem[32768];   // 64 KB
  // phase1: bufA[2] @ 0/8192 (128x64), bufB[2] @ 16384/24576 (128x64)
  // phase2: att @ 0 (128x128), UT @ 16384 (128x128, byte addr 32768)

  const int b    = blockIdx.y;
  const int X0   = blockIdx.x * TILE_X;
  const int tid  = threadIdx.x;
  const int wv   = tid >> 5;
  const int lane = tid & 31;
  const int half = lane >> 4;
  const int l16  = lane & 15;

  const _Float16* hb16  = h16 + (size_t)(b * JX_N + X0) * D_N;
  const _Float16* B16b  = B16 + (size_t)b * JQ_N * D_N;
  const _Float16* u16Tb = u16T + (size_t)b * D_N * JQ_N;

  v8f acc[8];
  #pragma unroll
  for (int i = 0; i < 8; ++i)
    #pragma unroll
    for (int e = 0; e < 8; ++e) acc[i][e] = 0.0f;

  auto stageAB = [&](int c, int bi) {   // 8 async b128 ops / thread
    _Float16* bA = smem + bi * 8192;
    _Float16* bB = smem + 16384 + bi * 8192;
    const int kc = c * KC;
    #pragma unroll
    for (int it = 0; it < 4; ++it) {
      const int idx = tid + 256 * it;           // 0..1023
      const int r = idx >> 3, g = idx & 7;      // row 0..127, 16B group 0..7
      copy16(hb16 + (size_t)r * D_N + kc + g * 8, bA + r * KC + g * 8);
      copy16(B16b + (size_t)r * D_N + kc + g * 8, bB + r * KC + g * 8);
    }
  };

  // ---- GEMM1: raw[x][q] = sum_d h16[x,d] * B16[q,d] (s_h folded in) ----
  stageAB(0, 0);
  for (int c = 0; c < D_N / KC; ++c) {          // 8 chunks
    const int cur = c & 1;
    if (c < 7) {
      stageAB(c + 1, cur ^ 1);                  // overlaps WMMA of chunk c
      WAIT_ASYNC(8);                            // chunk c landed (per wave)
    } else {
      WAIT_ASYNC(0);
    }
    if (c < 6)                                  // L2 warm for chunk c+2
      __builtin_prefetch(hb16 + (size_t)(wv * 16 + l16) * D_N + (c + 2) * KC, 0, 1);
    __syncthreads();                            // everyone's chunk c visible

    const _Float16* ldsAc = smem + cur * 8192;
    const _Float16* ldsBc = smem + 16384 + cur * 8192;
    #pragma unroll
    for (int kb = 0; kb < KC; kb += 32) {
      Frag af;
      const _Float16* arow = ldsAc + (wv * 16 + l16) * KC + kb;
      #pragma unroll
      for (int j = 0; j < 8; ++j)
        af.u[j] = *reinterpret_cast<const unsigned int*>(arow + a_k0(j, half));
      #pragma unroll
      for (int nt = 0; nt < 8; ++nt) {
        Frag bf;
        const _Float16* brow = ldsBc + (nt * 16 + l16) * KC + kb;
        #pragma unroll
        for (int v = 0; v < 8; ++v)
          bf.u[v] = *reinterpret_cast<const unsigned int*>(brow + a_k0(v, half));
        acc[nt] = __builtin_amdgcn_wmma_f32_16x16x32_f16(
            false, af.h, false, bf.h, (short)0, acc[nt], false, false);
      }
    }
    __syncthreads();                 // done reading before buffer is re-staged
  }

  // ---- logits = raw + su[q] + hmask ; softmax over q ; att -> LDS ----
  _Float16* att = smem;              // overwrites bufA region (barrier above)
  _Float16* UT  = smem + 16384;
  float suv[8], hmadd[8];
  #pragma unroll
  for (int nt = 0; nt < 8; ++nt) suv[nt] = su[b * JQ_N + nt * 16 + l16];
  #pragma unroll
  for (int v = 0; v < 8; ++v) {
    const int x = X0 + wv * 16 + half * 8 + v;
    hmadd[v] = hmask[b * JX_N + x] ? 0.0f : VERY_NEG;
  }
  #pragma unroll
  for (int nt = 0; nt < 8; ++nt)
    #pragma unroll
    for (int v = 0; v < 8; ++v) acc[nt][v] += suv[nt] + hmadd[v];

  #pragma unroll
  for (int v = 0; v < 8; ++v) {
    float m = -1e38f;
    #pragma unroll
    for (int nt = 0; nt < 8; ++nt) m = fmaxf(m, acc[nt][v]);
    #pragma unroll
    for (int s = 1; s < 16; s <<= 1) m = fmaxf(m, __shfl_xor(m, s, 32));
    float sum = 0.0f;
    #pragma unroll
    for (int nt = 0; nt < 8; ++nt) {
      const float e = __expf(acc[nt][v] - m);
      acc[nt][v] = e;
      sum += e;
    }
    #pragma unroll
    for (int s = 1; s < 16; s <<= 1) sum += __shfl_xor(sum, s, 32);
    const float inv = 1.0f / sum;
    const int xl = wv * 16 + half * 8 + v;
    if (l16 == 0) hlog[b * JX_N + X0 + xl] = m;
    #pragma unroll
    for (int nt = 0; nt < 8; ++nt)
      att[xl * 128 + nt * 16 + l16] = (_Float16)(acc[nt][v] * inv);
  }

  // ---- GEMM2: u_a[x][d] = sum_q att[x][q] * u16T[d][q], 4 chunks of 128 ----
  float* outb = out_ua + (size_t)(b * JX_N + X0) * D_N;
  for (int nc = 0; nc < 4; ++nc) {
    __syncthreads();                 // att ready / prev UT reads done
#if HAS_TDM
    if (wv == 0) {                   // one TDM descriptor stages the tile
      const unsigned long long ga =
          (unsigned long long)(uintptr_t)(u16Tb + (size_t)nc * 128 * JQ_N);
      u32x4 g0;
      g0.x = 1u;                                     // count=1 valid descriptor
      g0.y = 32768u;                                 // LDS byte addr of UT
      g0.z = (unsigned)ga;                           // global_addr[31:0]
      g0.w = (unsigned)((ga >> 32) & 0x1FFFFFFu) | (2u << 30);  // [56:32]|type=2
      i32x8 g1;
      g1[0] = (1 << 16);        // data_size = 2 bytes
      g1[1] = (128 << 16);      // tensor_dim0 = 128
      g1[2] = (128 << 16);      // tensor_dim1 = 128
      g1[3] = (128 << 16);      // tile_dim0 = 128
      g1[4] = 128;              // tile_dim1 = 128
      g1[5] = 128;              // tensor_dim0_stride = 128
      g1[6] = 0;
      g1[7] = 0;
      const i32x4 z4 = {0, 0, 0, 0};
#if __clang_major__ >= 23
      const i32x8 z8 = {0, 0, 0, 0, 0, 0, 0, 0};
      __builtin_amdgcn_tensor_load_to_lds(g0, g1, z4, z4, z8, 0);
#else
      __builtin_amdgcn_tensor_load_to_lds(g0, g1, z4, z4, 0);
#endif
      WAIT_TENSOR();
    }
#else
    #pragma unroll
    for (int it = 0; it < 8; ++it) {
      const int idx = tid + 256 * it;               // 0..2047
      const int n = idx >> 4, g = idx & 15;
      copy16(u16Tb + (size_t)(nc * 128 + n) * JQ_N + g * 8, UT + n * 128 + g * 8);
    }
    WAIT_ASYNC(0);
#endif
    __syncthreads();

    v8f acc2[8];
    #pragma unroll
    for (int i = 0; i < 8; ++i)
      #pragma unroll
      for (int e = 0; e < 8; ++e) acc2[i][e] = 0.0f;

    #pragma unroll
    for (int kb = 0; kb < JQ_N; kb += 32) {
      Frag af;
      const _Float16* arow = att + (wv * 16 + l16) * 128 + kb;
      #pragma unroll
      for (int j = 0; j < 8; ++j)
        af.u[j] = *reinterpret_cast<const unsigned int*>(arow + a_k0(j, half));
      #pragma unroll
      for (int nt = 0; nt < 8; ++nt) {
        Frag bf;
        const _Float16* brow = UT + (nt * 16 + l16) * 128 + kb;
        #pragma unroll
        for (int v = 0; v < 8; ++v)
          bf.u[v] = *reinterpret_cast<const unsigned int*>(brow + a_k0(v, half));
        acc2[nt] = __builtin_amdgcn_wmma_f32_16x16x32_f16(
            false, af.h, false, bf.h, (short)0, acc2[nt], false, false);
      }
    }
    #pragma unroll
    for (int nt = 0; nt < 8; ++nt)
      #pragma unroll
      for (int v = 0; v < 8; ++v)
        outb[(size_t)(wv * 16 + half * 8 + v) * D_N + nc * 128 + nt * 16 + l16] =
            acc2[nt][v];
  }
}

// ---------------------------------------------------------------------------
// hatt: per batch, softmax over x of h_logits, then h_a[d] = sum_x att_h*h16
// ---------------------------------------------------------------------------
__global__ __launch_bounds__(256)
void biattn_hatt(const _Float16* __restrict__ h16, const float* __restrict__ hlog,
                 float* __restrict__ ha) {
  __shared__ float attx[JX_N];
  __shared__ float red[256];
  const int b = blockIdx.x;
  const int tid = threadIdx.x;
  float m = -1e38f;
  for (int x = tid; x < JX_N; x += 256) m = fmaxf(m, hlog[b * JX_N + x]);
  red[tid] = m; __syncthreads();
  for (int s = 128; s > 0; s >>= 1) {
    if (tid < s) red[tid] = fmaxf(red[tid], red[tid + s]);
    __syncthreads();
  }
  m = red[0]; __syncthreads();
  float sum = 0.0f;
  for (int x = tid; x < JX_N; x += 256) {
    const float e = __expf(hlog[b * JX_N + x] - m);
    attx[x] = e;
    sum += e;
  }
  red[tid] = sum; __syncthreads();
  for (int s = 128; s > 0; s >>= 1) {
    if (tid < s) red[tid] += red[tid + s];
    __syncthreads();
  }
  const float inv = 1.0f / red[0];
  __syncthreads();
  const _Float16* hb = h16 + (size_t)b * JX_N * D_N;
  float a0 = 0.0f, a1 = 0.0f;
  for (int x = 0; x < JX_N; ++x) {
    const float w = attx[x];
    a0 += w * (float)hb[(size_t)x * D_N + tid];
    a1 += w * (float)hb[(size_t)x * D_N + tid + 256];
  }
  ha[b * D_N + tid] = a0 * inv;
  ha[b * D_N + tid + 256] = a1 * inv;
}

// ---------------------------------------------------------------------------
// bcast: out_ha[b,x,d] = ha[b,d]   (float4 stores)
// ---------------------------------------------------------------------------
__global__ __launch_bounds__(256)
void biattn_bcast(const float* __restrict__ ha, float* __restrict__ out_ha) {
  const size_t i = (size_t)blockIdx.x * blockDim.x + threadIdx.x;
  const size_t idx4 = i * 4;
  const int d = (int)(idx4 % D_N);
  const int b = (int)(idx4 / ((size_t)JX_N * D_N));
  const float4 v = *reinterpret_cast<const float4*>(ha + b * D_N + d);
  *reinterpret_cast<float4*>(out_ha + idx4) = v;
}

// ---------------------------------------------------------------------------
extern "C" void kernel_launch(void* const* d_in, const int* in_sizes, int n_in,
                              void* d_out, int out_size, void* d_ws, size_t ws_size,
                              hipStream_t stream) {
  const float* h = (const float*)d_in[0];
  const float* u = (const float*)d_in[1];
  const float* w = (const float*)d_in[2];
  const float* bias = (const float*)d_in[3];
  const unsigned char* h_mask = (const unsigned char*)d_in[4];
  const unsigned char* u_mask = (const unsigned char*)d_in[5];

  float* out_ua = (float*)d_out;
  float* out_ha = out_ua + (size_t)B_N * JX_N * D_N;

  float* wsf  = (float*)d_ws;
  float* su   = wsf;                          // 2048 floats
  float* hlog = wsf + 2048;                   // 16384 floats
  float* ha   = wsf + 2048 + 16384;           // 8192 floats
  _Float16* h16  = (_Float16*)(wsf + 26624);                 // 8M halfs
  _Float16* B16  = h16 + (size_t)B_N * JX_N * D_N;           // 1M halfs
  _Float16* u16T = B16 + (size_t)B_N * JQ_N * D_N;           // 1M halfs

  biattn_cvt_h<<<(B_N * JX_N * D_N / 8) / 256, 256, 0, stream>>>(h, h16);
  biattn_cvt_u<<<B_N, 256, 0, stream>>>(u, w, B16, u16T);
  biattn_prep<<<B_N, 256, 0, stream>>>(u, w, bias, u_mask, su);
  biattn_core<<<dim3(JX_N / TILE_X, B_N), 256, 0, stream>>>(h16, B16, u16T, h_mask,
                                                            su, out_ua, hlog);
  biattn_hatt<<<B_N, 256, 0, stream>>>(h16, hlog, ha);
  biattn_bcast<<<(B_N * JX_N * D_N / 4) / 256, 256, 0, stream>>>(ha, out_ha);
}
